// MultiheadLatentAttn_40724879901098
// MI455X (gfx1250) — compile-verified, matrix operand order checked by reading, not run on hardware
//
#include <hip/hip_runtime.h>
#include <hip/hip_bf16.h>

typedef __attribute__((ext_vector_type(16))) _Float16 v16h;
typedef __attribute__((ext_vector_type(8)))  _Float16 v8h;
typedef __attribute__((ext_vector_type(4)))  _Float16 v4h;
typedef __attribute__((ext_vector_type(8)))  float    v8f;

#define BB 4
#define SS 2048
#define DD 1024
#define HH 16
#define DQK 20
#define DV 64
#define DLQ 256
#define DLKV 128
#define SCALE_QK 0.22360679774997896f  /* 20^-0.5, folded into W_eff */

__device__ __forceinline__ v8f zero8() {
  v8f z; for (int i = 0; i < 8; ++i) z[i] = 0.0f; return z;
}

// D = A(16x32 f16) * B(32x16 f16) + C, f32 accumulate.
__device__ __forceinline__ v8f wmma16(v16h a, v16h b, v8f c) {
  return __builtin_amdgcn_wmma_f32_16x16x32_f16(false, a, false, b, (short)0, c,
                                                false, false);
}

// Load one 16-bit WMMA operand fragment from a [row][k] (row-major, ld halfs)
// tile. Per ISA 7.12.2: lane L holds row (L&15); K-halves split at (L>>4)*8
// with pairing {kb..kb+7, kb+16..kb+23}. Two 16B loads per lane.
__device__ __forceinline__ v16h load_frag(const _Float16* __restrict__ p, int ld) {
  const int lane = threadIdx.x & 31;
  const _Float16* q = p + (size_t)(lane & 15) * ld + ((lane >> 4) << 3);
  union { v16h v; v8h h[2]; } u;
  u.h[0] = *(const v8h*)(q);
  u.h[1] = *(const v8h*)(q + 16);
  return u.v;
}
__device__ __forceinline__ v16h load_frag_lds(const _Float16* p, int ld) {
  const int lane = threadIdx.x & 31;
  const _Float16* q = p + (size_t)(lane & 15) * ld + ((lane >> 4) << 3);
  union { v16h v; v8h h[2]; } u;
  u.h[0] = *(const v8h*)(q);
  u.h[1] = *(const v8h*)(q + 16);
  return u.v;
}

__device__ __forceinline__ v4h cvt4(float4 f) {
  v4h hv;
  hv[0] = (_Float16)f.x; hv[1] = (_Float16)f.y;
  hv[2] = (_Float16)f.z; hv[3] = (_Float16)f.w;
  return hv;
}

// ---------------- precompute kernels ----------------

__global__ void k_cvt(const float* __restrict__ wq, const float* __restrict__ wkv,
                      _Float16* __restrict__ wA) {
  int i = blockIdx.x * 256 + threadIdx.x;           // 384*1024 total
  if (i < 256 * 1024) wA[i] = (_Float16)wq[i];
  else                wA[i] = (_Float16)wkv[i - 256 * 1024];
}

// W_eff[n=(h*128+c)][k<256] = sum_q wq_b[h*20+q][k] * wkv_b[h][q][c], *scale
__global__ void k_weff(const float* __restrict__ wq_b, const float* __restrict__ wq_bb,
                       const float* __restrict__ wkv_b,
                       _Float16* __restrict__ Weff, float* __restrict__ beff) {
  int idx = blockIdx.x * 256 + threadIdx.x;         // 2048*256
  int n = idx >> 8, k = idx & 255;
  int hh = n >> 7, c = n & 127;
  float s = 0.0f;
  for (int q = 0; q < DQK; ++q)
    s += wq_b[(size_t)(hh * DQK + q) * DLQ + k] *
         wkv_b[(size_t)(hh * (DQK + DV) + q) * DLKV + c];
  Weff[idx] = (_Float16)(s * SCALE_QK);
  if (k == 0) {
    float bb = 0.0f;
    for (int q = 0; q < DQK; ++q)
      bb += wq_bb[hh * DQK + q] *
            wkv_b[(size_t)(hh * (DQK + DV) + q) * DLKV + c];
    beff[n] = bb * SCALE_QK;
  }
}

// W2[n<1024][k=(h*128+c)] = sum_v wkv_b[h][20+v][c] * wo[n][h*64+v]
__global__ void k_w2(const float* __restrict__ wkv_b, const float* __restrict__ wo,
                     _Float16* __restrict__ W2) {
  int idx = blockIdx.x * 256 + threadIdx.x;         // 1024*2048
  int n = idx >> 11, k = idx & 2047;
  int hh = k >> 7, c = k & 127;
  float s = 0.0f;
  for (int v = 0; v < DV; ++v)
    s += wkv_b[(size_t)(hh * (DQK + DV) + DQK + v) * DLKV + c] *
         wo[(size_t)n * DD + hh * DV + v];
  W2[idx] = (_Float16)s;
}

// ---------------- fused latent projection + RMSNorm ----------------
// Per block: 16 tokens x 384 outputs (q-lat 256 ++ kv-lat 128), K=1024.
// Ping-pong pipelined: LDS A tile double-buffered (one barrier per chunk),
// h loads 2 chunks ahead, B fragments 1 chunk ahead, no rotation copies.
__global__ __launch_bounds__(128) void k_latproj(
    const float* __restrict__ h, const _Float16* __restrict__ wA,
    const float* __restrict__ bq, const float* __restrict__ bkv,
    const float* __restrict__ qw, const float* __restrict__ kw,
    _Float16* __restrict__ qlat,   // [8192][256]
    _Float16* __restrict__ kvn,    // [B][S][128]
    _Float16* __restrict__ kvnT) { // [B][128][S]
  __shared__ __align__(16) _Float16 aT[2][16][40];
  __shared__ float ctile[16][392];
  __shared__ float part[16][16];
  __shared__ float inv[2][16];
  const int m0 = blockIdx.x * 16;
  const int wave = threadIdx.x >> 5, lane = threadIdx.x & 31;
  const int sr = threadIdx.x >> 3;          // 0..15 token row
  const int sc4 = (threadIdx.x & 7) * 4;    // 0..28 column base
  const int nb = wave * 96;                 // wave's N base (6 tiles of 16)
  const float* hrow = h + (size_t)(m0 + sr) * DD + sc4;
  const _Float16* wb = wA + (size_t)nb * DD;

  v8f acc[6];
#pragma unroll
  for (int t = 0; t < 6; ++t) acc[t] = zero8();

  // prologue: chunk 0 staged, chunk 1 in flight, B frags for chunk 0 in flight
  float4 f = *(const float4*)(hrow);
  *(v4h*)(&aT[0][sr][sc4]) = cvt4(f);
  f = *(const float4*)(hrow + 32);
  v16h bc[6], bn[6];
#pragma unroll
  for (int j = 0; j < 6; ++j) bc[j] = load_frag(wb + (size_t)j * 16 * DD, DD);
  __syncthreads();

  for (int c = 0; c < 32; c += 2) {
    // ---- even: consume aT[0] + bc; stage chunk c+1 -> aT[1]; load bn @ c+1
    *(v4h*)(&aT[1][sr][sc4]) = cvt4(f);
    if (c + 2 < 32) f = *(const float4*)(hrow + (c + 2) * 32);
#pragma unroll
    for (int j = 0; j < 6; ++j)
      bn[j] = load_frag(wb + (size_t)j * 16 * DD + (c + 1) * 32, DD);
    {
      v16h af = load_frag_lds(&aT[0][0][0], 40);
#pragma unroll
      for (int j = 0; j < 6; ++j) acc[j] = wmma16(af, bc[j], acc[j]);
    }
    __syncthreads();
    // ---- odd: consume aT[1] + bn; stage chunk c+2 -> aT[0]; load bc @ c+2
    if (c + 2 < 32) {
      *(v4h*)(&aT[0][sr][sc4]) = cvt4(f);
      if (c + 3 < 32) f = *(const float4*)(hrow + (c + 3) * 32);
#pragma unroll
      for (int j = 0; j < 6; ++j)
        bc[j] = load_frag(wb + (size_t)j * 16 * DD + (c + 2) * 32, DD);
    }
    {
      v16h af = load_frag_lds(&aT[1][0][0], 40);
#pragma unroll
      for (int j = 0; j < 6; ++j) acc[j] = wmma16(af, bn[j], acc[j]);
    }
    __syncthreads();
  }

  for (int t = 0; t < 6; ++t) {
    int n = nb + t * 16 + (lane & 15);
    float bias = (n < DLQ) ? bq[n] : bkv[n - DLQ];
    for (int e = 0; e < 8; ++e)
      ctile[e + ((lane >> 4) << 3)][n] = acc[t][e] + bias;
  }
  __syncthreads();
  {
    int r = threadIdx.x >> 3, p = threadIdx.x & 7;
    float sq = 0.0f, sk = 0.0f;
    for (int c = p * 32; c < p * 32 + 32; ++c) sq += ctile[r][c] * ctile[r][c];
    for (int c = DLQ + p * 16; c < DLQ + p * 16 + 16; ++c) sk += ctile[r][c] * ctile[r][c];
    part[r][p] = sq; part[r][8 + p] = sk;
  }
  __syncthreads();
  if (threadIdx.x < 16) {
    int r = threadIdx.x;
    float sq = 0.0f, sk = 0.0f;
    for (int p = 0; p < 8; ++p) { sq += part[r][p]; sk += part[r][8 + p]; }
    inv[0][r] = rsqrtf(sq * (1.0f / DLQ) + 1e-6f);
    inv[1][r] = rsqrtf(sk * (1.0f / DLKV) + 1e-6f);
  }
  __syncthreads();
  const int b = m0 >> 11, s0 = m0 & (SS - 1);
  for (int j = 0; j < 3; ++j) {
    int c = threadIdx.x + j * 128;
    if (c < DLQ) {
      float w = qw[c];
      for (int r = 0; r < 16; ++r)
        qlat[(size_t)(m0 + r) * DLQ + c] = (_Float16)(ctile[r][c] * inv[0][r] * w);
    } else {
      int cc = c - DLQ;
      float w = kw[cc];
      for (int r = 0; r < 16; ++r) {
        _Float16 hv = (_Float16)(ctile[r][c] * inv[1][r] * w);
        kvn[((size_t)b * SS + (s0 + r)) * DLKV + cc] = hv;
        kvnT[((size_t)b * DLKV + cc) * SS + (s0 + r)] = hv;
      }
    }
  }
}

// Ping-pong GEMM inner-loop helpers (no rotation copies).
#define LOAD_SET(AF, B0, B1, B2, B3, OFF, LD)         \
  AF = load_frag(pa + (OFF), LD);                     \
  B0 = load_frag(pb + (OFF), LD);                     \
  B1 = load_frag(pb + 16 * (LD) + (OFF), LD);         \
  B2 = load_frag(pb + 32 * (LD) + (OFF), LD);         \
  B3 = load_frag(pb + 48 * (LD) + (OFF), LD);
#define WMMA_SET(AF, B0, B1, B2, B3)                  \
  acc[0] = wmma16(AF, B0, acc[0]);                    \
  acc[1] = wmma16(AF, B1, acc[1]);                    \
  acc[2] = wmma16(AF, B2, acc[2]);                    \
  acc[3] = wmma16(AF, B3, acc[3]);

// ---------------- absorbed-Q GEMM: q_kv = q_lat @ W_effT ----------------
__global__ __launch_bounds__(256) void k_qkv(
    const _Float16* __restrict__ A, const _Float16* __restrict__ Bm,
    const float* __restrict__ bias, _Float16* __restrict__ qkv) {
  const int wave = threadIdx.x >> 5, lane = threadIdx.x & 31;
  const int m0 = blockIdx.x * 32 + (wave >> 2) * 16;
  const int n0 = blockIdx.y * 256 + (wave & 3) * 64;
  const _Float16* pa = A + (size_t)m0 * DLQ;
  const _Float16* pb = Bm + (size_t)n0 * DLQ;
  v8f acc[4];
#pragma unroll
  for (int t = 0; t < 4; ++t) acc[t] = zero8();
  v16h a0, a1, b00, b01, b02, b03, b10, b11, b12, b13;
  LOAD_SET(a0, b00, b01, b02, b03, 0, DLQ);
  int kc = 0;
#pragma unroll
  for (; kc + 64 < DLQ; kc += 64) {
    LOAD_SET(a1, b10, b11, b12, b13, kc + 32, DLQ);
    WMMA_SET(a0, b00, b01, b02, b03);
    LOAD_SET(a0, b00, b01, b02, b03, kc + 64, DLQ);
    WMMA_SET(a1, b10, b11, b12, b13);
  }
  LOAD_SET(a1, b10, b11, b12, b13, kc + 32, DLQ);
  WMMA_SET(a0, b00, b01, b02, b03);
  WMMA_SET(a1, b10, b11, b12, b13);
  for (int t = 0; t < 4; ++t) {
    int n = n0 + t * 16 + (lane & 15);
    int hh = n >> 7, c = n & 127;
    float bv = bias[n];
    for (int e = 0; e < 8; ++e) {
      int m = m0 + e + ((lane >> 4) << 3);
      int b = m >> 11, s = m & (SS - 1);
      qkv[(((size_t)b * HH + hh) * SS + s) * DLKV + c] = (_Float16)(acc[t][e] + bv);
    }
  }
}

// ---------------- flash attention over latent kv ----------------
__global__ __launch_bounds__(128) void k_attn(
    const _Float16* __restrict__ qkv,  // [B][H][S][128]
    const _Float16* __restrict__ kvn,  // [B][S][128]
    const _Float16* __restrict__ kvnT, // [B][128][S]
    _Float16* __restrict__ vlat) {     // [B][S][H*128]
  __shared__ __align__(16) _Float16 plds[4][16][40];
  const int wave = threadIdx.x >> 5, lane = threadIdx.x & 31;
  const int b = blockIdx.z, hh = blockIdx.y;
  const int q0 = (blockIdx.x * 4 + wave) * 16;
  const int hi8 = (lane >> 4) << 3, nLoc = lane & 15;

  const _Float16* qbase = qkv + (((size_t)b * HH + hh) * SS + q0) * DLKV;
  v16h afrag[4];
#pragma unroll
  for (int kc = 0; kc < 4; ++kc) afrag[kc] = load_frag(qbase + kc * 32, DLKV);

  v8f acc[8];
#pragma unroll
  for (int t = 0; t < 8; ++t) acc[t] = zero8();
  float mrow[8], lrow[8];
#pragma unroll
  for (int e = 0; e < 8; ++e) { mrow[e] = -1e30f; lrow[e] = 0.0f; }

  const int lastChunk = (q0 + 15) >> 5;
  for (int ch = 0; ch <= lastChunk; ++ch) {
    const int t0 = ch * 32;
    // scores: preload all 8 B fragments, then two independent wmma chains
    const _Float16* kb0 = kvn + ((size_t)b * SS + t0) * DLKV;
    const _Float16* kb1 = kb0 + 16 * DLKV;
    v16h kf0[4], kf1[4];
#pragma unroll
    for (int kc = 0; kc < 4; ++kc) {
      kf0[kc] = load_frag(kb0 + kc * 32, DLKV);
      kf1[kc] = load_frag(kb1 + kc * 32, DLKV);
    }
    v8f sc[2]; sc[0] = zero8(); sc[1] = zero8();
#pragma unroll
    for (int kc = 0; kc < 4; ++kc) sc[0] = wmma16(afrag[kc], kf0[kc], sc[0]);
#pragma unroll
    for (int kc = 0; kc < 4; ++kc) sc[1] = wmma16(afrag[kc], kf1[kc], sc[1]);

    // hoist P.V B-fragment loads so they overlap the softmax VALU work
    const _Float16* vb = kvnT + (size_t)b * DLKV * SS + t0;   // [c][t]
    v16h vf[8];
#pragma unroll
    for (int ct = 0; ct < 8; ++ct) vf[ct] = load_frag(vb + (size_t)(ct * 16) * SS, SS);

    if (t0 + 31 > q0) {              // causal mask only near diagonal (uniform branch)
#pragma unroll
      for (int half = 0; half < 2; ++half) {
        int t = t0 + half * 16 + nLoc;
#pragma unroll
        for (int e = 0; e < 8; ++e)
          if (t > q0 + e + hi8) sc[half][e] = -1e30f;
      }
    }
    float corrv[8];
#pragma unroll
    for (int e = 0; e < 8; ++e) {
      float mx = fmaxf(sc[0][e], sc[1][e]);
      for (int off = 8; off >= 1; off >>= 1) mx = fmaxf(mx, __shfl_xor(mx, off, 32));
      float mnew = fmaxf(mrow[e], mx);
      float corr = __expf(mrow[e] - mnew);
      float p0 = __expf(sc[0][e] - mnew);
      float p1 = __expf(sc[1][e] - mnew);
      sc[0][e] = p0; sc[1][e] = p1;
      float ps = p0 + p1;
      for (int off = 8; off >= 1; off >>= 1) ps += __shfl_xor(ps, off, 32);
      lrow[e] = lrow[e] * corr + ps;
      mrow[e] = mnew;
      corrv[e] = corr;
    }
#pragma unroll
    for (int t = 0; t < 8; ++t)
#pragma unroll
      for (int e = 0; e < 8; ++e) acc[t][e] *= corrv[e];
    // P tile -> LDS -> A fragment (same-wave LDS RAW; DS ops are in-order)
#pragma unroll
    for (int half = 0; half < 2; ++half)
#pragma unroll
      for (int e = 0; e < 8; ++e)
        plds[wave][e + hi8][half * 16 + nLoc] = (_Float16)sc[half][e];
    asm volatile("s_wait_dscnt 0x0" ::: "memory");
    v16h pf = load_frag_lds(&plds[wave][0][0], 40);
#pragma unroll
    for (int ct = 0; ct < 8; ++ct) acc[ct] = wmma16(pf, vf[ct], acc[ct]);
  }
  float linv[8];
#pragma unroll
  for (int e = 0; e < 8; ++e) linv[e] = 1.0f / lrow[e];
#pragma unroll
  for (int ct = 0; ct < 8; ++ct) {
    int c = hh * DLKV + ct * 16 + nLoc;
#pragma unroll
    for (int e = 0; e < 8; ++e) {
      int s = q0 + e + hi8;
      vlat[((size_t)b * SS + s) * (HH * DLKV) + c] = (_Float16)(acc[ct][e] * linv[e]);
    }
  }
}

// ---------------- absorbed output GEMM: out = v_lat @ W2T + wo_b ----------------
__global__ __launch_bounds__(256) void k_out(
    const _Float16* __restrict__ A, const _Float16* __restrict__ Bm,
    const float* __restrict__ bias, float* __restrict__ out) {
  const int KK = HH * DLKV;
  const int wave = threadIdx.x >> 5, lane = threadIdx.x & 31;
  const int m0 = blockIdx.x * 32 + (wave >> 2) * 16;
  const int n0 = blockIdx.y * 256 + (wave & 3) * 64;
  const _Float16* pa = A + (size_t)m0 * KK;
  const _Float16* pb = Bm + (size_t)n0 * KK;
  v8f acc[4];
#pragma unroll
  for (int t = 0; t < 4; ++t) acc[t] = zero8();
  v16h a0, a1, b00, b01, b02, b03, b10, b11, b12, b13;
  LOAD_SET(a0, b00, b01, b02, b03, 0, KK);
  int kc = 0;
  for (; kc + 64 < KK; kc += 64) {
    LOAD_SET(a1, b10, b11, b12, b13, kc + 32, KK);
    WMMA_SET(a0, b00, b01, b02, b03);
    LOAD_SET(a0, b00, b01, b02, b03, kc + 64, KK);
    WMMA_SET(a1, b10, b11, b12, b13);
  }
  LOAD_SET(a1, b10, b11, b12, b13, kc + 32, KK);
  WMMA_SET(a0, b00, b01, b02, b03);
  WMMA_SET(a1, b10, b11, b12, b13);
  for (int t = 0; t < 4; ++t) {
    int n = n0 + t * 16 + (lane & 15);
    float bv = bias[n];
    for (int e = 0; e < 8; ++e) {
      int m = m0 + e + ((lane >> 4) << 3);
      out[(size_t)m * DD + n] = acc[t][e] + bv;
    }
  }
}

extern "C" void kernel_launch(void* const* d_in, const int* in_sizes, int n_in,
                              void* d_out, int out_size, void* d_ws, size_t ws_size,
                              hipStream_t stream) {
  const float* h        = (const float*)d_in[0];
  /* d_in[1] = mask: causal handled analytically */
  const float* wq_a_w   = (const float*)d_in[2];
  const float* wq_a_b   = (const float*)d_in[3];
  const float* q_norm_w = (const float*)d_in[4];
  const float* wq_b_w   = (const float*)d_in[5];
  const float* wq_b_b   = (const float*)d_in[6];
  const float* wkv_a_w  = (const float*)d_in[7];
  const float* wkv_a_b  = (const float*)d_in[8];
  const float* kv_norm_w= (const float*)d_in[9];
  const float* wkv_b_w  = (const float*)d_in[10];
  const float* wo_w     = (const float*)d_in[11];
  const float* wo_b     = (const float*)d_in[12];
  float* out = (float*)d_out;

  char* p = (char*)d_ws;
  auto take = [&](size_t bytes) {
    char* r = p; p += (bytes + 255) & ~(size_t)255; return r;
  };
  _Float16* wA   = (_Float16*)take((size_t)384 * DD * 2);
  _Float16* Weff = (_Float16*)take((size_t)HH * DLKV * DLQ * 2);
  float*    beff = (float*)   take((size_t)HH * DLKV * 4);
  _Float16* W2   = (_Float16*)take((size_t)DD * HH * DLKV * 2);
  _Float16* qlat = (_Float16*)take((size_t)BB * SS * DLQ * 2);
  _Float16* kvn  = (_Float16*)take((size_t)BB * SS * DLKV * 2);
  _Float16* kvnT = (_Float16*)take((size_t)BB * DLKV * SS * 2);
  _Float16* qkv  = (_Float16*)take((size_t)BB * HH * SS * DLKV * 2);
  _Float16* vlat = (_Float16*)take((size_t)BB * SS * HH * DLKV * 2);
  (void)ws_size; (void)n_in; (void)in_sizes; (void)out_size;

  k_cvt    <<<1536, 256, 0, stream>>>(wq_a_w, wkv_a_w, wA);
  k_weff   <<<2048, 256, 0, stream>>>(wq_b_w, wq_b_b, wkv_b_w, Weff, beff);
  k_w2     <<<8192, 256, 0, stream>>>(wkv_b_w, wo_w, W2);
  k_latproj<<<(BB * SS) / 16, 128, 0, stream>>>(h, wA, wq_a_b, wkv_a_b,
                                                q_norm_w, kv_norm_w, qlat, kvn, kvnT);
  k_qkv    <<<dim3((BB * SS) / 32, (HH * DLKV) / 256), 256, 0, stream>>>(qlat, Weff, beff, qkv);
  k_attn   <<<dim3(SS / 64, HH, BB), 128, 0, stream>>>(qkv, kvn, kvnT, vlat);
  k_out    <<<dim3((BB * SS) / 32, DD / 256), 256, 0, stream>>>(vlat, W2, wo_b, out);
}
